// MultiHeadAttentionPooling_16673063043579
// MI455X (gfx1250) — matrix operand order repaired and checked by reading, not compiled, not run on hardware
//
#include <hip/hip_runtime.h>
#include <hip/hip_bf16.h>

#define B_      32
#define S_      4096
#define E_      1024
#define H_      16
#define D_      64
#define Q_      8
#define CHUNKS_ 8
#define CROWS_  (S_/CHUNKS_)    // 512 rows per chunk
#define STILE_  32              // S rows per iteration
#define NIT_    (CROWS_/STILE_) // 16 iterations
#define PSTR_   (8+8+Q_*D_)     // 528 floats per (b,h,chunk) partial
#define SCALE_  0.125f          // 64^-0.5 / temperature
#define NEGBIG  (-3.4e38f)

typedef __attribute__((ext_vector_type(16))) __bf16 v16bf;
typedef __attribute__((ext_vector_type(2)))  __bf16 v2bf;
typedef __attribute__((ext_vector_type(8)))  float  v8f;

struct U32x8 { uint4 a, b; };

__device__ __forceinline__ unsigned short f2bf(float f) {
  unsigned u = __builtin_bit_cast(unsigned, f);
  unsigned r = u + 0x7FFFu + ((u >> 16) & 1u);   // round-to-nearest-even
  return (unsigned short)(r >> 16);
}

__device__ __forceinline__ unsigned pack_bf2(float a, float b) {
#if __has_builtin(__builtin_amdgcn_cvt_pk_bf16_f32)
  v2bf p = __builtin_amdgcn_cvt_pk_bf16_f32(a, b);
  return __builtin_bit_cast(unsigned, p);
#else
  return (unsigned)f2bf(a) | ((unsigned)f2bf(b) << 16);
#endif
}

// Build a 16 x bf16 WMMA fragment from two 16-byte chunks.
__device__ __forceinline__ v16bf ld_frag(const unsigned short* p0, const unsigned short* p1) {
  U32x8 t;
  t.a = *(const uint4*)p0;
  t.b = *(const uint4*)p1;
  return __builtin_bit_cast(v16bf, t);
}

// ---------------------------------------------------------------------------
// Prep 0: x (B*S*E f32) -> bf16 copy in workspace (for the async DMA path)
// ---------------------------------------------------------------------------
__global__ __launch_bounds__(256) void xbf_kernel(const float* __restrict__ x,
                                                  unsigned short* __restrict__ xbf) {
  const size_t i = ((size_t)blockIdx.x * 256 + threadIdx.x) * 8;
  const float4 v0 = *(const float4*)(x + i);
  const float4 v1 = *(const float4*)(x + i + 4);
  uint2 p0, p1;
  p0.x = pack_bf2(v0.x, v0.y); p0.y = pack_bf2(v0.z, v0.w);
  p1.x = pack_bf2(v1.x, v1.y); p1.y = pack_bf2(v1.z, v1.w);
  *(uint2*)(xbf + i)     = p0;
  *(uint2*)(xbf + i + 4) = p1;
}

// ---------------------------------------------------------------------------
// Prep 1: Wkv (1024 x 2048 f32, row-major over K) -> Wkv_t bf16 [n][k]
// ---------------------------------------------------------------------------
__global__ __launch_bounds__(256) void wkvt_kernel(const float* __restrict__ Wkv,
                                                   unsigned short* __restrict__ wkvt) {
  const int n = blockIdx.x;                  // 0..2047
  for (int k = 2 * threadIdx.x; k < E_; k += 512) {
    const float a = Wkv[(size_t)k * (2 * E_) + n];
    const float b = Wkv[(size_t)(k + 1) * (2 * E_) + n];
    *(unsigned*)(wkvt + (size_t)n * E_ + k) = pack_bf2(a, b);
  }
}

// ---------------------------------------------------------------------------
// Prep 2: q = LayerNorm(queries) @ Wq, per-head, rows padded 8->16, bf16
// ---------------------------------------------------------------------------
__global__ __launch_bounds__(256) void qproj_kernel(const float* __restrict__ queries,
                                                    const float* __restrict__ gamma,
                                                    const float* __restrict__ beta,
                                                    const float* __restrict__ Wq,
                                                    unsigned short* __restrict__ qb) {
  __shared__ float qn[Q_][E_];
  __shared__ float red[256];
  const int h = blockIdx.x;
  const int t = threadIdx.x;

  for (int r = 0; r < Q_; ++r) {
    float s = 0.f, s2 = 0.f;
    for (int e = t; e < E_; e += 256) {
      float v = queries[r * E_ + e];
      s += v; s2 += v * v;
    }
    red[t] = s; __syncthreads();
    for (int o = 128; o > 0; o >>= 1) { if (t < o) red[t] += red[t + o]; __syncthreads(); }
    float mu = red[0] * (1.f / E_); __syncthreads();
    red[t] = s2; __syncthreads();
    for (int o = 128; o > 0; o >>= 1) { if (t < o) red[t] += red[t + o]; __syncthreads(); }
    float var = red[0] * (1.f / E_) - mu * mu; __syncthreads();
    float rstd = rsqrtf(var + 1e-5f);
    for (int e = t; e < E_; e += 256)
      qn[r][e] = (queries[r * E_ + e] - mu) * rstd * gamma[e] + beta[e];
    __syncthreads();
  }

  for (int p = 0; p < 2; ++p) {
    const int r = p * 4 + (t >> 6);
    const int d = t & 63;
    float s = 0.f;
    for (int e = 0; e < E_; ++e)
      s += qn[r][e] * Wq[(size_t)e * E_ + h * D_ + d];
    qb[(h * 16 + r) * D_ + d] = f2bf(s);
  }
  for (int i = t; i < 8 * D_; i += 256)
    qb[(h * 16 + 8) * D_ + i] = 0;
}

// ---------------------------------------------------------------------------
// Main: fused KV projection + flash-attention pooling over one (batch, chunk).
// 256 threads = 8 waves; each wave handles 2 heads.
// ASYNC=1: double-buffered global_load_async_to_lds of bf16 x tiles.
// ASYNC=0: direct f32 loads + in-kernel conversion (small-workspace fallback).
// ---------------------------------------------------------------------------
template <int ASYNC>
__global__ __launch_bounds__(256) void attn_kernel(const float* __restrict__ xg,
                                                   const unsigned short* __restrict__ xbf,
                                                   const unsigned char* __restrict__ kpm,
                                                   const unsigned short* __restrict__ wkvt,
                                                   const unsigned short* __restrict__ qb,
                                                   float* __restrict__ part) {
  extern __shared__ unsigned short smem[];
  const int t    = threadIdx.x;
  const int lane = t & 31;
  const int wave = t >> 5;
  const int ln15 = lane & 15;
  const int hi   = lane >> 4;
  const int b    = blockIdx.x / CHUNKS_;
  const int c    = blockIdx.x % CHUNKS_;

  // LDS layout (ushort units): tile buffer(s), then per-wave scratch
  unsigned short* wsc = smem + (ASYNC ? 2 : 1) * 32768 + wave * 4608;
  unsigned short* ksd = wsc;                     // k  [s:32][d:64]
  unsigned short* vtd = wsc + 32 * D_;           // vT [d:64][s:32]
  unsigned short* pls = wsc + 32 * D_ + D_ * 32; // P  [q:16][s:32]

  // Issue one 64KB bf16 tile copy (256B per thread) with async global->LDS DMA.
  auto issue_tile = [&](int nit, int buf) {
    const char* g = (const char*)(xbf + ((size_t)b * S_ + c * CROWS_ + nit * STILE_) * E_)
                    + (size_t)t * 16;
    unsigned loff = (unsigned)(buf * 65536 + t * 16);
#pragma unroll
    for (int i = 0; i < 16; ++i) {
      asm volatile("global_load_async_to_lds_b128 %0, %1, off"
                   :: "v"(loff), "v"(g) : "memory");
      loff += 4096;
      g    += 4096;
    }
  };

  // Query A-fragments (16x32 bf16 each; K = head_dim halves)
  v16bf qa[2][2];
#pragma unroll
  for (int hh = 0; hh < 2; ++hh) {
    const int h = wave * 2 + hh;
#pragma unroll
    for (int f = 0; f < 2; ++f) {
      const unsigned short* qp = qb + (h * 16 + ln15) * D_ + f * 32 + hi * 8;
      qa[hh][f] = ld_frag(qp, qp + 16);
    }
  }

  float m[2][8], l[2][8];
  v8f acc[2][4];
#pragma unroll
  for (int hh = 0; hh < 2; ++hh) {
#pragma unroll
    for (int r = 0; r < 8; ++r) { m[hh][r] = NEGBIG; l[hh][r] = 0.f; }
#pragma unroll
    for (int n = 0; n < 4; ++n)
#pragma unroll
      for (int j = 0; j < 8; ++j) acc[hh][n][j] = 0.f;
  }

  if constexpr (ASYNC) issue_tile(0, 0);

  for (int it = 0; it < NIT_; ++it) {
    const int s0 = c * CROWS_ + it * STILE_;
    unsigned short* xt;

    if constexpr (ASYNC) {
      asm volatile("s_wait_asynccnt 0" ::: "memory"); // tile for this iter landed
      __syncthreads();                                 // visible to all waves
      if (it + 1 < NIT_) issue_tile(it + 1, (it & 1) ^ 1);
      xt = smem + (it & 1) * 32768;
    } else {
      __syncthreads();
#pragma unroll 4
      for (int i = 0; i < 32; ++i) {
        const int idx = i * 256 + t;
        const int row = idx >> 8;
        const int c4  = idx & 255;
        const float4 v = *(const float4*)(xg + ((size_t)b * S_ + s0 + row) * E_ + c4 * 4);
        uint2 pk;
        pk.x = pack_bf2(v.x, v.y);
        pk.y = pack_bf2(v.z, v.w);
        *(uint2*)(smem + row * E_ + c4 * 4) = pk;
      }
      __syncthreads();
      if (it + 1 < NIT_)
        __builtin_prefetch(xg + ((size_t)b * S_ + s0 + STILE_ + (t >> 3)) * E_ + (t & 7) * 128, 0, 3);
      xt = smem;
    }

    bool msk[2];
#pragma unroll
    for (int st = 0; st < 2; ++st)
      msk[st] = kpm[(size_t)b * S_ + s0 + st * 16 + ln15] != 0;

#pragma unroll
    for (int hh = 0; hh < 2; ++hh) {
      const int h = wave * 2 + hh;

      // ---- KV projection, register-blocked: per kv hold [ms:2][nt:4] C tiles,
      //      6 fragment loads per 8 WMMAs ----
#pragma unroll
      for (int kv = 0; kv < 2; ++kv) {
        v8f C[2][4];
#pragma unroll
        for (int ms = 0; ms < 2; ++ms)
#pragma unroll
          for (int nt = 0; nt < 4; ++nt)
#pragma unroll
            for (int j = 0; j < 8; ++j) C[ms][nt][j] = 0.f;

#pragma unroll 2
        for (int ks = 0; ks < 32; ++ks) {
          const unsigned short* a0p = xt + ln15 * E_ + ks * 32 + hi * 8;
          const unsigned short* a1p = a0p + 16 * E_;
          const v16bf A0 = ld_frag(a0p, a0p + 16);
          const v16bf A1 = ld_frag(a1p, a1p + 16);
#pragma unroll
          for (int nt = 0; nt < 4; ++nt) {
            const unsigned short* bp =
                wkvt + (size_t)(kv * E_ + h * D_ + nt * 16 + ln15) * E_ + ks * 32 + hi * 8;
            const v16bf Bf = ld_frag(bp, bp + 16);
            C[0][nt] = __builtin_amdgcn_wmma_f32_16x16x32_bf16(false, A0, false, Bf,
                                                               (short)0, C[0][nt], false, false);
            C[1][nt] = __builtin_amdgcn_wmma_f32_16x16x32_bf16(false, A1, false, Bf,
                                                               (short)0, C[1][nt], false, false);
          }
        }

        // store: k as [s][d] (scattered b16), v as [d][s] (contiguous b128)
#pragma unroll
        for (int ms = 0; ms < 2; ++ms)
#pragma unroll
          for (int nt = 0; nt < 4; ++nt) {
            if (kv == 0) {
#pragma unroll
              for (int r = 0; r < 8; ++r) {
                const int sl = ms * 16 + r + hi * 8;
                ksd[sl * D_ + nt * 16 + ln15] = f2bf(C[ms][nt][r]);
              }
            } else {
              const int dl = nt * 16 + ln15;
              uint4 pk;
              pk.x = pack_bf2(C[ms][nt][0], C[ms][nt][1]);
              pk.y = pack_bf2(C[ms][nt][2], C[ms][nt][3]);
              pk.z = pack_bf2(C[ms][nt][4], C[ms][nt][5]);
              pk.w = pack_bf2(C[ms][nt][6], C[ms][nt][7]);
              *(uint4*)(vtd + dl * STILE_ + ms * 16 + hi * 8) = pk;
            }
          }
      }

      // ---- scores = q(16x64) @ k^T(64x32) ----
      v8f cs[2];
#pragma unroll
      for (int st = 0; st < 2; ++st) {
#pragma unroll
        for (int j = 0; j < 8; ++j) cs[st][j] = 0.f;
#pragma unroll
        for (int kk = 0; kk < 2; ++kk) {
          const unsigned short* bp = ksd + (st * 16 + ln15) * D_ + kk * 32 + hi * 8;
          const v16bf bb = ld_frag(bp, bp + 16);
          cs[st] = __builtin_amdgcn_wmma_f32_16x16x32_bf16(false, qa[hh][kk], false, bb,
                                                           (short)0, cs[st], false, false);
        }
#pragma unroll
        for (int j = 0; j < 8; ++j)
          cs[st][j] = msk[st] ? NEGBIG : cs[st][j] * SCALE_;
      }

      // ---- online softmax per q-row ----
      float cr[8];
#pragma unroll
      for (int r = 0; r < 8; ++r) {
        float v0 = cs[0][r], v1 = cs[1][r];
        float rm = fmaxf(v0, v1);
#pragma unroll
        for (int o = 8; o >= 1; o >>= 1) rm = fmaxf(rm, __shfl_xor(rm, o, 32));
        const float mn = fmaxf(m[hh][r], rm);
        const float p0 = (v0 > -1e30f) ? __expf(v0 - mn) : 0.f;
        const float p1 = (v1 > -1e30f) ? __expf(v1 - mn) : 0.f;
        float rs = p0 + p1;
#pragma unroll
        for (int o = 8; o >= 1; o >>= 1) rs += __shfl_xor(rs, o, 32);
        const float co = (m[hh][r] > -1e30f) ? __expf(m[hh][r] - mn) : 0.f;
        cr[r] = co;
        l[hh][r] = l[hh][r] * co + rs;
        m[hh][r] = mn;
        const int qrow = r + hi * 8;
        pls[qrow * STILE_ + ln15]      = f2bf(p0);
        pls[qrow * STILE_ + 16 + ln15] = f2bf(p1);
      }
#pragma unroll
      for (int n = 0; n < 4; ++n)
#pragma unroll
        for (int r = 0; r < 8; ++r) acc[hh][n][r] *= cr[r];

      // ---- PV: acc(16x64) += P(16x32) @ v(32x64) ----
      const unsigned short* pp = pls + ln15 * STILE_ + hi * 8;
      const v16bf pa = ld_frag(pp, pp + 16);
#pragma unroll
      for (int n = 0; n < 4; ++n) {
        const unsigned short* bp = vtd + (n * 16 + ln15) * STILE_ + hi * 8;
        const v16bf bb = ld_frag(bp, bp + 16);
        acc[hh][n] = __builtin_amdgcn_wmma_f32_16x16x32_bf16(false, pa, false, bb,
                                                             (short)0, acc[hh][n], false, false);
      }
    } // hh
  } // it

  // ---- write partials ----
#pragma unroll
  for (int hh = 0; hh < 2; ++hh) {
    const int h = wave * 2 + hh;
    float* pb = part + (((size_t)b * H_ + h) * CHUNKS_ + c) * PSTR_;
    if (lane < 16) {
#pragma unroll
      for (int r = 0; r < 8; ++r)
#pragma unroll
        for (int n = 0; n < 4; ++n)
          pb[16 + r * D_ + n * 16 + lane] = acc[hh][n][r];
      if (lane == 0) {
#pragma unroll
        for (int r = 0; r < 8; ++r) { pb[r] = m[hh][r]; pb[8 + r] = l[hh][r]; }
      }
    }
  }
}

// ---------------------------------------------------------------------------
// Reduce chunks -> pooled(b,q,:), then out = pooled @ Wo
// ---------------------------------------------------------------------------
__global__ __launch_bounds__(256) void reduce_kernel(const float* __restrict__ part,
                                                     const float* __restrict__ Wo,
                                                     float* __restrict__ out) {
  __shared__ float pooled[E_];
  __shared__ float wgt[H_][CHUNKS_];
  const int t = threadIdx.x;
  const int b = blockIdx.x / Q_;
  const int q = blockIdx.x % Q_;

  if (t < H_) {
    const int h = t;
    const float* pb = part + ((size_t)b * H_ + h) * CHUNKS_ * PSTR_;
    float M = NEGBIG;
    for (int c = 0; c < CHUNKS_; ++c) M = fmaxf(M, pb[c * PSTR_ + q]);
    float w[CHUNKS_]; float L = 0.f;
    for (int c = 0; c < CHUNKS_; ++c) {
      const float mc = pb[c * PSTR_ + q];
      const float e = (mc > -1e30f) ? __expf(mc - M) : 0.f;
      w[c] = e;
      L += pb[c * PSTR_ + 8 + q] * e;
    }
    const float invL = (L > 0.f) ? 1.f / L : 0.f;
    for (int c = 0; c < CHUNKS_; ++c) wgt[h][c] = w[c] * invL;
  }
  __syncthreads();

  for (int j = t; j < E_; j += 256) {
    const int h = j >> 6, d = j & 63;
    const float* pb = part + ((size_t)b * H_ + h) * CHUNKS_ * PSTR_;
    float s = 0.f;
    for (int c = 0; c < CHUNKS_; ++c) s += wgt[h][c] * pb[c * PSTR_ + 16 + q * D_ + d];
    pooled[j] = s;
  }
  __syncthreads();

  for (int e = t; e < E_; e += 256) {
    float s = 0.f;
    for (int k = 0; k < E_; ++k) s += pooled[k] * Wo[(size_t)k * E_ + e];
    out[((size_t)b * Q_ + q) * E_ + e] = s;
  }
}

// ---------------------------------------------------------------------------
extern "C" void kernel_launch(void* const* d_in, const int* in_sizes, int n_in,
                              void* d_out, int out_size, void* d_ws, size_t ws_size,
                              hipStream_t stream) {
  (void)in_sizes; (void)n_in; (void)out_size;
  const float*         x       = (const float*)d_in[0];
  const unsigned char* kpm     = (const unsigned char*)d_in[1]; // bool mask, 1B/elem
  const float*         queries = (const float*)d_in[2];
  const float*         gamma   = (const float*)d_in[3];
  const float*         beta    = (const float*)d_in[4];
  const float*         Wq      = (const float*)d_in[5];
  const float*         Wkv     = (const float*)d_in[6];
  const float*         Wo      = (const float*)d_in[7];
  float*               out     = (float*)d_out;

  // workspace layout
  const size_t XBF_BYTES  = (size_t)B_ * S_ * E_ * 2;          // 256 MiB (async path)
  const size_t WKVT_BYTES = (size_t)2 * E_ * E_ * 2;           // 4 MiB
  const size_t QB_BYTES   = (size_t)H_ * 16 * D_ * 2;          // 32 KiB
  const size_t PART_BYTES = (size_t)B_ * H_ * CHUNKS_ * PSTR_ * 4;
  const bool use_async = ws_size >= XBF_BYTES + WKVT_BYTES + QB_BYTES + PART_BYTES;

  char* wsp = (char*)d_ws;
  unsigned short* xbf  = (unsigned short*)wsp;
  char* tail = wsp + (use_async ? XBF_BYTES : 0);
  unsigned short* wkvt = (unsigned short*)tail;
  unsigned short* qb   = (unsigned short*)(tail + WKVT_BYTES);
  float*          part = (float*)(tail + WKVT_BYTES + QB_BYTES);

  wkvt_kernel<<<2 * E_, 256, 0, stream>>>(Wkv, wkvt);
  qproj_kernel<<<H_, 256, 0, stream>>>(queries, gamma, beta, Wq, qb);

  if (use_async) {
    xbf_kernel<<<(B_ * S_ * E_) / (256 * 8), 256, 0, stream>>>(x, xbf);
    const size_t lds_bytes = 2 * 65536 + 8 * 9216;  // 200 KB
    attn_kernel<1><<<B_ * CHUNKS_, 256, lds_bytes, stream>>>(x, xbf, kpm, wkvt, qb, part);
  } else {
    const size_t lds_bytes = 65536 + 8 * 9216;      // 136 KB
    attn_kernel<0><<<B_ * CHUNKS_, 256, lds_bytes, stream>>>(x, xbf, kpm, wkvt, qb, part);
  }

  reduce_kernel<<<B_ * Q_, 256, 0, stream>>>(part, Wo, out);
}
